// SpanConsistencyNetwork_19026705121554
// MI455X (gfx1250) — compile-verified, matrix operand order checked by reading, not run on hardware
//
#include <hip/hip_runtime.h>
#include <hip/hip_bf16.h>
#include <math.h>

// ---------------------------------------------------------------------------
// Problem constants (from reference)
// ---------------------------------------------------------------------------
#define BB 16
#define NN 2048
#define HH 1024
#define D1 512
#define D2 256
#define MM 5
#define MAXW 15
#define NEG_INF_F (-10000.0f)
#define NTOK (BB * NN)          // 32768 tokens
#define TOK_TILE 32             // tokens per block
#define KC 32                   // K chunk for wmma 16x16x32
#define KCP 40                  // padded K stride (80 B: 16B-aligned, bank-friendly)
#define TS1 (D1 + 8)            // padded T1 stride (1040 B)

typedef __attribute__((ext_vector_type(16))) _Float16 v16h;
typedef __attribute__((ext_vector_type(8)))  float    v8f;

__device__ __forceinline__ float gelu_erf(float x) {
    return 0.5f * x * (1.0f + erff(x * 0.70710678118654752f));
}
__device__ __forceinline__ float sigmoidf(float x) {
    return 1.0f / (1.0f + __expf(-x));
}

// CDNA5 async global->LDS copy (ASYNCcnt-tracked DMA, no VGPR round trip).
// VDST = VGPR holding LDS byte offset, VADDR = 64-bit global address.
__device__ __forceinline__ void async_copy_b128(unsigned lds_byte_off, const void* gptr) {
    asm volatile("global_load_async_to_lds_b128 %0, %1, off"
                 :: "v"(lds_byte_off), "v"(gptr)
                 : "memory");
}
__device__ __forceinline__ void wait_async0() {
    asm volatile("s_wait_asynccnt 0x0" ::: "memory");
}

// ---------------------------------------------------------------------------
// Kernel 0: one-time fp32 -> f16 convert + transpose of W1, W2 into workspace.
//   W1t[m][col][k] (col-major over output dim -> K-contiguous B tiles)
// ---------------------------------------------------------------------------
__global__ __launch_bounds__(256) void cvt_weights_kernel(
    const float* __restrict__ W1, const float* __restrict__ W2,
    _Float16* __restrict__ W1t, _Float16* __restrict__ W2t)
{
    const size_t NW1 = (size_t)MM * D1 * HH;
    const size_t NW2 = (size_t)MM * D2 * D1;
    const size_t o = (size_t)blockIdx.x * 256 + threadIdx.x;
    if (o < NW1) {
        const size_t m = o / ((size_t)D1 * HH);
        const size_t r = o % ((size_t)D1 * HH);
        const size_t c = r / HH, k = r % HH;
        W1t[o] = (_Float16)W1[(m * HH + k) * D1 + c];
    } else if (o < NW1 + NW2) {
        const size_t p = o - NW1;
        const size_t m = p / ((size_t)D2 * D1);
        const size_t r = p % ((size_t)D2 * D1);
        const size_t c = r / D1, k = r % D1;
        W2t[p] = (_Float16)W2[(m * D1 + k) * D2 + c];
    }
}

// ---------------------------------------------------------------------------
// Kernel 1: fused per-marker token scorer (WMMA f16, f32 accumulate).
//   t1 = gelu(h @ W1[m] + b1[m])   [32 x 512]
//   t2 = gelu(t1 @ W2[m] + b2[m])  [32 x 256]
//   scores[m, tok] = t2 . W3[m] + b3[m]
// grid = (NTOK/TOK_TILE, M), block = 256 (8 wave32)
// ---------------------------------------------------------------------------
__global__ __launch_bounds__(256) void scorer_wmma_kernel(
    const float* __restrict__ h,
    const _Float16* __restrict__ W1t, const float* __restrict__ b1,
    const _Float16* __restrict__ W2t, const float* __restrict__ b2,
    const float* __restrict__ W3, const float* __restrict__ b3,
    float* __restrict__ scores)
{
    __shared__ _Float16 sA [TOK_TILE * KCP];  //  2.5 KB : h chunk (f16, padded)
    __shared__ _Float16 sBt[128 * KCP];       // 10.0 KB : W^T chunk (K-contig per col)
    __shared__ _Float16 sT1[TOK_TILE * TS1];  // 32.5 KB : gelu(h@W1)
    __shared__ _Float16 sT2[TOK_TILE * D2];   // 16.0 KB : gelu(t1@W2)

    const int m       = blockIdx.y;
    const int tokBase = blockIdx.x * TOK_TILE;
    const int tid     = threadIdx.x;
    const int lane    = tid & 31;
    const int wave    = tid >> 5;            // 0..7 -> 16-col tile within 128-col pass
    const int l15     = lane & 15;
    const int halfk   = lane >> 4;           // 0/1 : K-half per ISA fragment layout

    const _Float16* __restrict__ W1tm = W1t + (size_t)m * D1 * HH;
    const _Float16* __restrict__ W2tm = W2t + (size_t)m * D2 * D1;

    // ---------------- Stage 1: T1 = gelu(h @ W1m + b1) ----------------
    for (int pass = 0; pass < D1 / 128; ++pass) {
        const int colBase = pass * 128;
        v8f acc0 = {};   // rows 0..15  of token tile
        v8f acc1 = {};   // rows 16..31 of token tile

        for (int kc = 0; kc < HH; kc += KC) {
            __syncthreads();   // previous chunk fully consumed before DMA overwrite
            // async DMA the B^T tile: 128 cols x 32 K f16 = 8 KB (16 B per lane x2)
            #pragma unroll
            for (int j = 0; j < 2; ++j) {
                const int idx  = tid + j * 256;    // 0..511
                const int col  = idx >> 2;         // 0..127
                const int part = idx & 3;          // 4 x 16B chunks per column
                async_copy_b128((unsigned)(uintptr_t)&sBt[col * KCP + part * 8],
                                &W1tm[(size_t)(colBase + col) * HH + kc + part * 8]);
            }
            // stage A chunk: 32 tokens x 32 K, fp32 -> f16 (one b128 load/thread)
            {
                const int r  = tid >> 3;
                const int kb = (tid & 7) * 4;
                const float4 hv = *reinterpret_cast<const float4*>(
                    &h[(size_t)(tokBase + r) * HH + kc + kb]);
                _Float16* dst = &sA[r * KCP + kb];
                dst[0] = (_Float16)hv.x; dst[1] = (_Float16)hv.y;
                dst[2] = (_Float16)hv.z; dst[3] = (_Float16)hv.w;
            }
            wait_async0();
            __syncthreads();

            v16h a0, a1, bf;
            #pragma unroll
            for (int v = 0; v < 8; ++v) {
                const int k = (v & 3) * 2 + (v >> 2) * 16 + halfk * 8;
                a0[2 * v]     = sA[l15 * KCP + k];
                a0[2 * v + 1] = sA[l15 * KCP + k + 1];
                a1[2 * v]     = sA[(l15 + 16) * KCP + k];
                a1[2 * v + 1] = sA[(l15 + 16) * KCP + k + 1];
                bf[2 * v]     = sBt[(wave * 16 + l15) * KCP + k];
                bf[2 * v + 1] = sBt[(wave * 16 + l15) * KCP + k + 1];
            }
            acc0 = __builtin_amdgcn_wmma_f32_16x16x32_f16(false, a0, false, bf,
                                                          (short)0, acc0, false, false);
            acc1 = __builtin_amdgcn_wmma_f32_16x16x32_f16(false, a1, false, bf,
                                                          (short)0, acc1, false, false);
        }
        // epilogue: bias + gelu -> sT1 (f16)
        const int col  = colBase + wave * 16 + l15;           // 0..511
        const float bv = b1[m * D1 + col];
        #pragma unroll
        for (int v = 0; v < 8; ++v) {
            const int row = v + halfk * 8;                    // C/D VGPR layout
            sT1[row * TS1 + col]        = (_Float16)gelu_erf(acc0[v] + bv);
            sT1[(row + 16) * TS1 + col] = (_Float16)gelu_erf(acc1[v] + bv);
        }
    }

    // ---------------- Stage 2: T2 = gelu(T1 @ W2m + b2) ----------------
    for (int pass = 0; pass < D2 / 128; ++pass) {
        const int colBase = pass * 128;
        v8f acc0 = {};
        v8f acc1 = {};
        for (int kc = 0; kc < D1; kc += KC) {
            __syncthreads();
            #pragma unroll
            for (int j = 0; j < 2; ++j) {
                const int idx  = tid + j * 256;
                const int col  = idx >> 2;
                const int part = idx & 3;
                async_copy_b128((unsigned)(uintptr_t)&sBt[col * KCP + part * 8],
                                &W2tm[(size_t)(colBase + col) * D1 + kc + part * 8]);
            }
            wait_async0();
            __syncthreads();

            v16h a0, a1, bf;
            #pragma unroll
            for (int v = 0; v < 8; ++v) {
                const int k = (v & 3) * 2 + (v >> 2) * 16 + halfk * 8;
                a0[2 * v]     = sT1[l15 * TS1 + kc + k];
                a0[2 * v + 1] = sT1[l15 * TS1 + kc + k + 1];
                a1[2 * v]     = sT1[(l15 + 16) * TS1 + kc + k];
                a1[2 * v + 1] = sT1[(l15 + 16) * TS1 + kc + k + 1];
                bf[2 * v]     = sBt[(wave * 16 + l15) * KCP + k];
                bf[2 * v + 1] = sBt[(wave * 16 + l15) * KCP + k + 1];
            }
            acc0 = __builtin_amdgcn_wmma_f32_16x16x32_f16(false, a0, false, bf,
                                                          (short)0, acc0, false, false);
            acc1 = __builtin_amdgcn_wmma_f32_16x16x32_f16(false, a1, false, bf,
                                                          (short)0, acc1, false, false);
        }
        const int col  = colBase + wave * 16 + l15;           // 0..255
        const float bv = b2[m * D2 + col];
        #pragma unroll
        for (int v = 0; v < 8; ++v) {
            const int row = v + halfk * 8;
            sT2[row * D2 + col]        = (_Float16)gelu_erf(acc0[v] + bv);
            sT2[(row + 16) * D2 + col] = (_Float16)gelu_erf(acc1[v] + bv);
        }
    }

    __syncthreads();
    // ---------------- Stage 3: scores = T2 . W3 + b3 ----------------
    {
        const int tok = tid >> 3;                 // 32 tokens x 8 threads
        const int sub = tid & 7;
        const float* __restrict__ W3m = W3 + m * D2;
        float p = 0.0f;
        for (int e = sub * 32; e < sub * 32 + 32; ++e)
            p += (float)sT2[tok * D2 + e] * W3m[e];
        p += __shfl_xor(p, 4, 8);
        p += __shfl_xor(p, 2, 8);
        p += __shfl_xor(p, 1, 8);
        if (sub == 0)
            scores[(size_t)m * NTOK + tokBase + tok] = p + b3[m];
    }
}

// ---------------------------------------------------------------------------
// Kernel 2: span-consistency boost. grid = (B, M), block = 256.
// ---------------------------------------------------------------------------
__global__ __launch_bounds__(256) void boost_kernel(
    const int*   __restrict__ mask,
    const float* __restrict__ gamma,
    const float* __restrict__ width_logits,
    const float* __restrict__ scores,
    float*       __restrict__ boosted)
{
    __shared__ float sS[NN];     // masked scores
    __shared__ float sMin[NN];   // running width-w sliding min
    __shared__ float sBoost[NN]; // running max over widths

    const int b   = blockIdx.x;
    const int m   = blockIdx.y;
    const int tid = threadIdx.x;

    float wl[MAXW], ww[MAXW];
    float mx = -1e30f;
    #pragma unroll
    for (int i = 0; i < MAXW; ++i) { wl[i] = width_logits[m * MAXW + i]; mx = fmaxf(mx, wl[i]); }
    float sum = 0.0f;
    #pragma unroll
    for (int i = 0; i < MAXW; ++i) { ww[i] = __expf(wl[i] - mx); sum += ww[i]; }
    const float inv = 1.0f / sum;
    #pragma unroll
    for (int i = 0; i < MAXW; ++i) ww[i] *= inv;

    const float* __restrict__ srow = scores + (size_t)m * NTOK + (size_t)b * NN;
    const int*   __restrict__ mrow = mask + (size_t)b * NN;

    for (int i = tid; i < NN; i += 256) {
        const float s  = srow[i];
        const float ms = (mrow[i] == 0) ? NEG_INF_F : s;
        sS[i] = ms; sMin[i] = ms; sBoost[i] = ms;      // width-1 term
    }
    __syncthreads();

    for (int w = 2; w <= MAXW; ++w) {
        for (int i = tid; i <= NN - w; i += 256)
            sMin[i] = fminf(sMin[i], sS[i + w - 1]);
        __syncthreads();
        const float wwv = ww[w - 2];
        for (int i = tid; i < NN; i += 256) {
            const int j0 = (i - w + 1 > 0) ? (i - w + 1) : 0;
            const int j1 = (i < NN - w) ? i : (NN - w);
            float e = -INFINITY;
            for (int j = j0; j <= j1; ++j) e = fmaxf(e, sMin[j] * wwv);
            sBoost[i] = fmaxf(sBoost[i], e);
        }
        __syncthreads();
    }

    const float g = gamma[m];
    for (int i = tid; i < NN; i += 256)
        boosted[(size_t)m * NTOK + (size_t)b * NN + i] = srow[i] + g * sBoost[i];
}

// ---------------------------------------------------------------------------
// Kernel 3: cross-marker refinement; one thread per token.
// ---------------------------------------------------------------------------
__global__ __launch_bounds__(256) void refine_kernel(
    const float* __restrict__ boosted,
    const float* __restrict__ corr,
    const float* __restrict__ cw1, const float* __restrict__ cb1,
    const float* __restrict__ cw2, const float* __restrict__ cb2,
    const float* __restrict__ gate,
    float* __restrict__ out)
{
    const int t = blockIdx.x * blockDim.x + threadIdx.x;
    if (t >= NTOK) return;

    float lg[MM], pr[MM], co[MM], rf[MM];
    #pragma unroll
    for (int m = 0; m < MM; ++m) {
        lg[m] = boosted[(size_t)m * NTOK + t];
        pr[m] = sigmoidf(lg[m]);
    }
    #pragma unroll
    for (int j = 0; j < MM; ++j) {
        float s = 0.0f;
        #pragma unroll
        for (int m = 0; m < MM; ++m) s += pr[m] * corr[m * MM + j];
        co[j] = s;
    }
    #pragma unroll
    for (int j = 0; j < MM; ++j) rf[j] = cb2[j];
    for (int k = 0; k < 64; ++k) {
        float hs = cb1[k];
        #pragma unroll
        for (int m = 0; m < MM; ++m) hs += co[m] * cw1[m * 64 + k];
        const float hg = gelu_erf(hs);
        #pragma unroll
        for (int j = 0; j < MM; ++j) rf[j] += hg * cw2[k * MM + j];
    }
    const float gt = sigmoidf(gate[0]);
    #pragma unroll
    for (int m = 0; m < MM; ++m)
        out[(size_t)t * MM + m] = lg[m] + gt * rf[m];
}

// ---------------------------------------------------------------------------
// Launch
// ---------------------------------------------------------------------------
extern "C" void kernel_launch(void* const* d_in, const int* in_sizes, int n_in,
                              void* d_out, int out_size, void* d_ws, size_t ws_size,
                              hipStream_t stream) {
    (void)in_sizes; (void)n_in; (void)out_size; (void)ws_size;
    const float* h     = (const float*)d_in[0];
    const int*   amask = (const int*)  d_in[1];
    const float* W1    = (const float*)d_in[2];
    const float* b1    = (const float*)d_in[3];
    const float* W2    = (const float*)d_in[4];
    const float* b2    = (const float*)d_in[5];
    const float* W3    = (const float*)d_in[6];
    const float* b3    = (const float*)d_in[7];
    const float* gamma = (const float*)d_in[8];
    const float* wlog  = (const float*)d_in[9];
    const float* corr  = (const float*)d_in[10];
    const float* cw1   = (const float*)d_in[11];
    const float* cb1   = (const float*)d_in[12];
    const float* cw2   = (const float*)d_in[13];
    const float* cb2   = (const float*)d_in[14];
    const float* gate  = (const float*)d_in[15];
    float*       out   = (float*)d_out;                          // [B, N, M] f32

    float*    ws_scores  = (float*)d_ws;                         // [M, NTOK] f32
    float*    ws_boosted = ws_scores + (size_t)MM * NTOK;        // [M, NTOK] f32
    _Float16* W1t        = (_Float16*)(ws_boosted + (size_t)MM * NTOK); // [M,512,1024] f16
    _Float16* W2t        = W1t + (size_t)MM * D1 * HH;                  // [M,256,512]  f16

    const size_t nW  = (size_t)MM * D1 * HH + (size_t)MM * D2 * D1;
    cvt_weights_kernel<<<dim3((nW + 255) / 256), 256, 0, stream>>>(W1, W2, W1t, W2t);
    scorer_wmma_kernel<<<dim3(NTOK / TOK_TILE, MM), 256, 0, stream>>>(
        h, W1t, b1, W2t, b2, W3, b3, ws_scores);
    boost_kernel<<<dim3(BB, MM), 256, 0, stream>>>(
        amask, gamma, wlog, ws_scores, ws_boosted);
    refine_kernel<<<dim3(NTOK / 256), 256, 0, stream>>>(
        ws_boosted, corr, cw1, cb1, cw2, cb2, gate, out);
}
